// DropoutWrapper_90357521973659
// MI455X (gfx1250) — 8-bit matrix operand order repaired and checked by reading, compiled, not run on hardware
//
#include <hip/hip_runtime.h>

typedef __attribute__((ext_vector_type(16))) int   v16i;
typedef __attribute__((ext_vector_type(8)))  float v8f;

#define BM 128
#define BN 128
#define BK 128

__global__ __launch_bounds__(256) void quant4_rows_fp8(
    const float* __restrict__ X, unsigned char* __restrict__ Q,
    float* __restrict__ S, int K) {
  const int row = blockIdx.x;
  const float* xr = X + (size_t)row * K;
  const int t = threadIdx.x;

  float vmin = 0.0f, vmax = 0.0f;
  for (int i = t * 8; i < K; i += 256 * 8) {
    const float4 a = *(const float4*)(xr + i);
    const float4 b = *(const float4*)(xr + i + 4);
    vmin = fminf(vmin, fminf(fminf(a.x, a.y), fminf(a.z, a.w)));
    vmin = fminf(vmin, fminf(fminf(b.x, b.y), fminf(b.z, b.w)));
    vmax = fmaxf(vmax, fmaxf(fmaxf(a.x, a.y), fmaxf(a.z, a.w)));
    vmax = fmaxf(vmax, fmaxf(fmaxf(b.x, b.y), fmaxf(b.z, b.w)));
  }

  __shared__ float smin[256], smax[256];
  smin[t] = vmin; smax[t] = vmax;
  __syncthreads();
  for (int s = 128; s > 0; s >>= 1) {
    if (t < s) {
      smin[t] = fminf(smin[t], smin[t + s]);
      smax[t] = fmaxf(smax[t], smax[t + s]);
    }
    __syncthreads();
  }

  __shared__ float sh_scale, sh_zero;
  if (t == 0) {
    float sc = (smax[0] - smin[0]) / 15.0f;
    if (sc == 0.0f) sc = 1.0f;
    sh_scale = sc;
    sh_zero  = rintf(-smin[0] / sc);
    S[row]   = sc;
  }
  __syncthreads();

  const float sc = sh_scale;
  const float z  = sh_zero;
  const int   zi = (int)z;

  for (int i = t * 8; i < K; i += 256 * 8) {
    const float4 a = *(const float4*)(xr + i);
    const float4 b = *(const float4*)(xr + i + 4);
    float vals[8] = {a.x, a.y, a.z, a.w, b.x, b.y, b.z, b.w};
    unsigned long long packed = 0ull;
#pragma unroll
    for (int e = 0; e < 8; e++) {
      float q = rintf(vals[e] / sc) + z;
      q = fminf(fmaxf(q, 0.0f), 15.0f);
      int p  = (int)q - zi;
      int ap = p < 0 ? -p : p;
      unsigned long long tab = (ap < 8) ? 0x4E4C4A4844403800ULL
                                        : 0x5756555453525150ULL;
      unsigned byte = (unsigned)((tab >> ((ap & 7) * 8)) & 0xFF);
      if (p < 0) byte |= 0x80u;
      packed |= (unsigned long long)byte << (e * 8);
    }
    *(unsigned long long*)(Q + (size_t)row * K + i) = packed;
  }
}

__global__ __launch_bounds__(256) void gemm_fp8_wmma(
    const unsigned char* __restrict__ qx, const unsigned char* __restrict__ qw,
    const float* __restrict__ sx, const float* __restrict__ sw,
    const float* __restrict__ bias, float* __restrict__ out,
    int M, int N, int K) {
  __shared__ __align__(16) unsigned char lA[2][BM * BK];
  __shared__ __align__(16) unsigned char lB[2][BN * BK];

  const int t    = threadIdx.x;
  const int lane = t & 31;
  const int wave = t >> 5;
  const int wm   = wave & 1;
  const int wn   = wave >> 1;

  const int m0 = blockIdx.y * BM;
  const int n0 = blockIdx.x * BN;

  const int lo16 = lane & 15;
  const int hi   = lane >> 4;

  v8f acc[4][2];
#pragma unroll
  for (int i = 0; i < 4; i++)
#pragma unroll
    for (int j = 0; j < 2; j++) {
      v8f zv = {0.f, 0.f, 0.f, 0.f, 0.f, 0.f, 0.f, 0.f};
      acc[i][j] = zv;
    }

  const int arow = t >> 3;
  const int aoff = (t & 7) * 16;

  auto issue_tile = [&](int buf, int ks) {
#pragma unroll
    for (int r = 0; r < 4; r++) {
      unsigned ldsa = (unsigned)(size_t)&lA[buf][r * 4096 + t * 16];
      const unsigned char* ga =
          qx + (size_t)(m0 + r * 32 + arow) * K + ks + aoff;
      asm volatile("global_load_async_to_lds_b128 %0, %1, off"
                   :: "v"(ldsa), "v"(ga) : "memory");
    }
#pragma unroll
    for (int r = 0; r < 4; r++) {
      unsigned ldsb = (unsigned)(size_t)&lB[buf][r * 4096 + t * 16];
      const unsigned char* gb =
          qw + (size_t)(n0 + r * 32 + arow) * K + ks + aoff;
      asm volatile("global_load_async_to_lds_b128 %0, %1, off"
                   :: "v"(ldsb), "v"(gb) : "memory");
    }
  };

  issue_tile(0, 0);

  int buf = 0;
  for (int ks = 0; ks < K; ks += BK) {
    asm volatile("s_wait_asynccnt 0x0" ::: "memory");
    __syncthreads();
    if (ks + BK < K) issue_tile(buf ^ 1, ks + BK);

    const unsigned char* Ab = lA[buf];
    const unsigned char* Bb = lB[buf];

    v16i bf[2];
#pragma unroll
    for (int j = 0; j < 2; j++) {
      const int col = wn * 32 + j * 16 + lo16;
      const unsigned char* base = Bb + col * BK + hi * 16;
#pragma unroll
      for (int c = 0; c < 4; c++) {
        uint4 q = *(const uint4*)(base + 32 * c);
        bf[j][4 * c + 0] = (int)q.x;
        bf[j][4 * c + 1] = (int)q.y;
        bf[j][4 * c + 2] = (int)q.z;
        bf[j][4 * c + 3] = (int)q.w;
      }
    }

#pragma unroll
    for (int i = 0; i < 4; i++) {
      const int rowm = wm * 64 + i * 16 + lo16;
      const unsigned char* base = Ab + rowm * BK + hi * 16;
      v16i af;
#pragma unroll
      for (int j2 = 0; j2 < 8; j2++) {
        unsigned long long d = *(const unsigned long long*)(base + 32 * (j2 >> 1) + 8 * (j2 & 1));
        af[2 * j2 + 0] = (int)(unsigned)d;
        af[2 * j2 + 1] = (int)(unsigned)(d >> 32);
      }
      acc[i][0] = __builtin_amdgcn_wmma_f32_16x16x128_fp8_fp8(
          af, bf[0], (short)0, acc[i][0], false, false);
      acc[i][1] = __builtin_amdgcn_wmma_f32_16x16x128_fp8_fp8(
          af, bf[1], (short)0, acc[i][1], false, false);
    }

    buf ^= 1;
  }

#pragma unroll
  for (int i = 0; i < 4; i++) {
#pragma unroll
    for (int j = 0; j < 2; j++) {
      const int gn  = n0 + wn * 32 + j * 16 + lo16;
      const float swv = sw[gn];
      const float bv  = bias[gn];
      const int   mb  = m0 + wm * 64 + i * 16 + hi * 8;
#pragma unroll
      for (int v = 0; v < 8; v++) {
        const int gm = mb + v;
        out[(size_t)gm * N + gn] = sx[gm] * swv * acc[i][j][v] + bv;
      }
    }
  }
}

extern "C" void kernel_launch(void* const* d_in, const int* in_sizes, int n_in,
                              void* d_out, int out_size, void* d_ws, size_t ws_size,
                              hipStream_t stream) {
  (void)n_in; (void)out_size; (void)ws_size;
  const float* x    = (const float*)d_in[0];
  const float* W    = (const float*)d_in[1];
  const float* bias = (const float*)d_in[2];
  float* out = (float*)d_out;

  const int N = in_sizes[2];
  const int K = in_sizes[1] / N;
  const int M = in_sizes[0] / K;

  unsigned char* qx = (unsigned char*)d_ws;
  unsigned char* qw = qx + (size_t)M * K;
  float* sx = (float*)(qw + (size_t)N * K);
  float* sw = sx + M;

  quant4_rows_fp8<<<M, 256, 0, stream>>>(x, qx, sx, K);
  quant4_rows_fp8<<<N, 256, 0, stream>>>(W, qw, sw, K);

  dim3 grid(N / BN, M / BM);
  gemm_fp8_wmma<<<grid, 256, 0, stream>>>(qx, qw, sx, sw, bias, out, M, N, K);
}
